// Intermediate_26156350832994
// MI455X (gfx1250) — matrix operand order repaired and checked by reading, not compiled, not run on hardware
//
#include <hip/hip_runtime.h>
#include <hip/hip_bf16.h>

// ---- problem constants (from reference) ----
#define M_ROWS 12800   // B*L = 128*100
#define H_DIM  256
#define F_DIM  1024    // 4*H
#define T_STEPS 4

typedef __attribute__((ext_vector_type(16))) __bf16 v16bf;
typedef __attribute__((ext_vector_type(8)))  float  v8f;

union ABf { v16bf v; unsigned short u[16]; uint4 q[2]; };

__device__ __forceinline__ unsigned short f2bf(float f) {
  union { float f; unsigned u; } cv; cv.f = f;
  unsigned u = cv.u;
  u += 0x7FFFu + ((u >> 16) & 1u);   // round-to-nearest-even
  return (unsigned short)(u >> 16);
}
__device__ __forceinline__ unsigned pack2bf(float a, float b) {
  return (unsigned)f2bf(a) | ((unsigned)f2bf(b) << 16);
}

// -------------------------------------------------------------------------
// Prep: fp32 -> bf16 weights (once per launch)
// -------------------------------------------------------------------------
__global__ void cvt_bf16_kernel(const float* __restrict__ src,
                                unsigned short* __restrict__ dst) {
  int i = (blockIdx.x * blockDim.x + threadIdx.x) * 4;
  float4 v = *(const float4*)(src + i);
  uint2 o;
  o.x = pack2bf(v.x, v.y);
  o.y = pack2bf(v.z, v.w);
  *(uint2*)(dst + i) = o;
}

// -------------------------------------------------------------------------
// Kernel 1: h = x @ W1^T + b1   (M x 1024), fused BN1 stats (sum, sumsq)
// grid: M/32 blocks, 512 threads (16 waves).
// wave w owns columns [w*64, w*64+64) (4 n-tiles) and both 16-row m-subtiles.
// acc = 4*2*8 = 64 VGPRs -> no spills.
// -------------------------------------------------------------------------
__global__ void gemm1_kernel(const float* __restrict__ x,
                             const unsigned short* __restrict__ W1bf,
                             const float* __restrict__ b1,
                             float* __restrict__ h, float* __restrict__ stats1) {
  __shared__ unsigned short lds_a[32][32];
  const int m0   = blockIdx.x * 32;
  const int tid  = threadIdx.x;
  const int wave = tid >> 5;
  const int lane = tid & 31;
  const int half = lane >> 4;
  const int ln16 = lane & 15;

  v8f acc[4][2];
  #pragma unroll
  for (int j = 0; j < 4; ++j)
    #pragma unroll
    for (int m = 0; m < 2; ++m)
      #pragma unroll
      for (int r = 0; r < 8; ++r) acc[j][m][r] = 0.f;

  #pragma unroll 1
  for (int kc = 0; kc < 8; ++kc) {            // K = 256 in chunks of 32
    __syncthreads();
    {  // cooperative fp32 -> bf16 staging of the 32x32 x-tile (2 elems/thread)
      int idx = tid * 2;
      int r = idx >> 5, c = idx & 31;
      float2 xv = *(const float2*)(&x[(size_t)(m0 + r) * H_DIM + kc * 32 + c]);
      *(unsigned*)&lds_a[r][c] = pack2bf(xv.x, xv.y);
    }
    __syncthreads();

    ABf afrag[2];
    #pragma unroll
    for (int e = 0; e < 16; ++e) {
      int k = half * 16 + e;  // contiguous k = 16*half + e (matches B)
      afrag[0].u[e] = lds_a[ln16][k];
      afrag[1].u[e] = lds_a[16 + ln16][k];
    }
    #pragma unroll
    for (int j = 0; j < 4; ++j) {
      int n = wave * 64 + j * 16 + ln16;
      const uint4* wp = (const uint4*)&W1bf[(size_t)n * H_DIM + kc * 32 + half * 16];
      ABf bfrag; bfrag.q[0] = wp[0]; bfrag.q[1] = wp[1];
      #pragma unroll
      for (int m = 0; m < 2; ++m)
        acc[j][m] = __builtin_amdgcn_wmma_f32_16x16x32_bf16(false, afrag[m].v, false, bfrag.v,
                                                            (short)0, acc[j][m], false, false);
    }
  }

  #pragma unroll
  for (int j = 0; j < 4; ++j) {
    int n = wave * 64 + j * 16 + ln16;
    float bias = b1[n];
    float s = 0.f, ss = 0.f;
    #pragma unroll
    for (int m = 0; m < 2; ++m) {
      #pragma unroll
      for (int r = 0; r < 8; ++r) {
        int mrow = m0 + m * 16 + half * 8 + r;   // ISA C/D-layout
        float v = acc[j][m][r] + bias;
        h[(size_t)mrow * F_DIM + n] = v;
        s += v; ss += v * v;
      }
    }
    atomicAdd(&stats1[n], s);
    atomicAdd(&stats1[F_DIM + n], ss);
  }
}

// -------------------------------------------------------------------------
// Finalize BN stats -> per-channel scale a, shift c
// -------------------------------------------------------------------------
__global__ void finalize_kernel(const float* __restrict__ stats,
                                const float* __restrict__ g, const float* __restrict__ be,
                                float* __restrict__ ab, int nch, float inv_count) {
  int c = blockIdx.x * blockDim.x + threadIdx.x;
  if (c >= nch) return;
  float mean = stats[c] * inv_count;
  float var  = stats[nch + c] * inv_count - mean * mean;
  float a = g[c] * rsqrtf(var + 1e-5f);
  ab[c] = a;
  ab[nch + c] = be[c] - mean * a;
}

// -------------------------------------------------------------------------
// Kernel 3: fused BN1 + constant-input LIF(4 steps) + GEMM2 + residual + BN2 stats
// h2[t,m,n] = sum_f s[t,m,f]*W2[n,f] + b2[n] + x[m,n]
// grid: M/32 blocks, 512 threads (16 waves).
// wave w owns columns [w*16, w*16+16) (1 n-tile), both m-subtiles, all 4 t.
// acc = 4*2*8 = 64 VGPRs; each B-frag feeds 8 WMMAs.
// -------------------------------------------------------------------------
__global__ void gemm2_kernel(const float* __restrict__ h,
                             const unsigned short* __restrict__ W2bf,
                             const float* __restrict__ b2, const float* __restrict__ x,
                             const float* __restrict__ ab1,
                             float* __restrict__ h2, float* __restrict__ stats2) {
  __shared__ unsigned short lds_s[T_STEPS][32][32];   // 8 KB of bf16 spikes
  const int m0   = blockIdx.x * 32;
  const int tid  = threadIdx.x;
  const int wave = tid >> 5;
  const int lane = tid & 31;
  const int half = lane >> 4;
  const int ln16 = lane & 15;

  v8f acc[T_STEPS][2];   // [t][msub]
  #pragma unroll
  for (int t = 0; t < T_STEPS; ++t)
    #pragma unroll
    for (int m = 0; m < 2; ++m)
      #pragma unroll
      for (int r = 0; r < 8; ++r) acc[t][m][r] = 0.f;

  #pragma unroll 1
  for (int kc = 0; kc < F_DIM / 32; ++kc) {
    __syncthreads();
    {  // cooperative: read h (float2), apply BN1, 4-step constant-input LIF -> spike LDS
      int idx = tid * 2;
      int r = idx >> 5, c = idx & 31;
      int f = kc * 32 + c;
      float2 hv = *(const float2*)(&h[(size_t)(m0 + r) * F_DIM + f]);
      float2 av = *(const float2*)(&ab1[f]);
      float2 cv = *(const float2*)(&ab1[F_DIM + f]);
      float z0 = av.x * hv.x + cv.x;
      float z1 = av.y * hv.y + cv.y;
      float v0 = 0.f, v1 = 0.f;
      #pragma unroll
      for (int t = 0; t < T_STEPS; ++t) {
        v0 = 0.5f * (v0 + z0);
        v1 = 0.5f * (v1 + z1);
        unsigned s0 = (v0 >= 1.f) ? 0x3F80u : 0u;     // bf16 1.0 / 0.0
        unsigned s1 = (v1 >= 1.f) ? 0x3F800000u : 0u; // already shifted to hi half
        *(unsigned*)&lds_s[t][r][c] = s0 | s1;
        if (v0 >= 1.f) v0 = 0.f;                      // hard reset
        if (v1 >= 1.f) v1 = 0.f;
      }
    }
    __syncthreads();

    ABf bfrag;
    {
      int n = wave * 16 + ln16;
      const uint4* wp = (const uint4*)&W2bf[(size_t)n * F_DIM + kc * 32 + half * 16];
      bfrag.q[0] = wp[0]; bfrag.q[1] = wp[1];
    }
    #pragma unroll
    for (int m = 0; m < 2; ++m) {
      ABf afrag[T_STEPS];
      #pragma unroll
      for (int e = 0; e < 16; ++e) {
        int k = half * 16 + e;
        #pragma unroll
        for (int t = 0; t < T_STEPS; ++t) afrag[t].u[e] = lds_s[t][m * 16 + ln16][k];
      }
      #pragma unroll
      for (int t = 0; t < T_STEPS; ++t)
        acc[t][m] = __builtin_amdgcn_wmma_f32_16x16x32_bf16(false, afrag[t].v, false,
                                                            bfrag.v, (short)0,
                                                            acc[t][m], false, false);
    }
  }

  {
    int n = wave * 16 + ln16;
    float bias = b2[n];
    float s = 0.f, ss = 0.f;
    #pragma unroll
    for (int m = 0; m < 2; ++m) {
      #pragma unroll
      for (int t = 0; t < T_STEPS; ++t) {
        #pragma unroll
        for (int r = 0; r < 8; ++r) {
          int mrow = m0 + m * 16 + half * 8 + r;
          float val = acc[t][m][r] + bias + x[(size_t)mrow * H_DIM + n];  // residual
          h2[((size_t)t * M_ROWS + mrow) * H_DIM + n] = val;
          s += val; ss += val * val;
        }
      }
    }
    atomicAdd(&stats2[n], s);
    atomicAdd(&stats2[H_DIM + n], ss);
  }
}

// -------------------------------------------------------------------------
// Kernel 5: BN2 + time-coupled LIF over t + mean over t; BN3 stats
// grid: 256 blocks x 256 threads; block handles 50 rows, thread = channel
// -------------------------------------------------------------------------
__global__ void lif2_kernel(const float* __restrict__ h2, const float* __restrict__ ab2,
                            float* __restrict__ out_pre, float* __restrict__ stats3) {
  const int n = threadIdx.x;
  const int mbase = blockIdx.x * 50;
  const float a = ab2[n], c = ab2[H_DIM + n];
  float s = 0.f, ss = 0.f;
  for (int i = 0; i < 50; ++i) {
    size_t idx = (size_t)(mbase + i) * H_DIM + n;
    float v = 0.f, sp = 0.f;
    #pragma unroll
    for (int t = 0; t < T_STEPS; ++t) {
      float z = a * h2[(size_t)t * M_ROWS * H_DIM + idx] + c;
      v = 0.5f * (v + z);
      if (v >= 1.f) { sp += 1.f; v = 0.f; }
    }
    float o = sp * 0.25f;
    out_pre[idx] = o;
    s += o; ss += o * o;
  }
  atomicAdd(&stats3[n], s);
  atomicAdd(&stats3[H_DIM + n], ss);
}

// -------------------------------------------------------------------------
// Kernel 7: apply BN3 -> output
// -------------------------------------------------------------------------
__global__ void final_kernel(const float* __restrict__ out_pre,
                             const float* __restrict__ ab3, float* __restrict__ out) {
  int idx = blockIdx.x * blockDim.x + threadIdx.x;
  int n = idx & (H_DIM - 1);
  out[idx] = ab3[n] * out_pre[idx] + ab3[H_DIM + n];
}

// -------------------------------------------------------------------------
extern "C" void kernel_launch(void* const* d_in, const int* in_sizes, int n_in,
                              void* d_out, int out_size, void* d_ws, size_t ws_size,
                              hipStream_t stream) {
  const float* x   = (const float*)d_in[0];
  const float* W1  = (const float*)d_in[1];
  const float* b1  = (const float*)d_in[2];
  const float* g1  = (const float*)d_in[3];
  const float* be1 = (const float*)d_in[4];
  const float* W2  = (const float*)d_in[5];
  const float* b2  = (const float*)d_in[6];
  const float* g2  = (const float*)d_in[7];
  const float* be2 = (const float*)d_in[8];
  const float* g3  = (const float*)d_in[9];
  const float* be3 = (const float*)d_in[10];

  float* W = (float*)d_ws;
  float* h      = W;                                      // 13107200 f
  float* h2     = h  + (size_t)M_ROWS * F_DIM;            // 13107200 f
  float* opre   = h2 + (size_t)T_STEPS * M_ROWS * H_DIM;  //  3276800 f
  float* stats1 = opre + (size_t)M_ROWS * H_DIM;          // 2*1024 f
  float* stats2 = stats1 + 2 * F_DIM;                     // 2*256 f
  float* stats3 = stats2 + 2 * H_DIM;                     // 2*256 f
  float* ab1    = stats3 + 2 * H_DIM;                     // 2*1024 f
  float* ab2    = ab1 + 2 * F_DIM;                        // 2*256 f
  float* ab3    = ab2 + 2 * H_DIM;                        // 2*256 f
  unsigned short* W1bf = (unsigned short*)(ab3 + 2 * H_DIM);        // 262144 bf16
  unsigned short* W2bf = W1bf + (size_t)F_DIM * H_DIM;              // 262144 bf16

  // zero all stats accumulators (contiguous region)
  hipMemsetAsync(stats1, 0, (2 * F_DIM + 4 * H_DIM) * sizeof(float), stream);

  // one-time weight conversion to bf16 (256K elements each, 4/thread)
  cvt_bf16_kernel<<<256, 256, 0, stream>>>(W1, W1bf);
  cvt_bf16_kernel<<<256, 256, 0, stream>>>(W2, W2bf);

  gemm1_kernel<<<M_ROWS / 32, 512, 0, stream>>>(x, W1bf, b1, h, stats1);
  finalize_kernel<<<(F_DIM + 255) / 256, 256, 0, stream>>>(stats1, g1, be1, ab1, F_DIM,
                                                           1.0f / (float)M_ROWS);
  gemm2_kernel<<<M_ROWS / 32, 512, 0, stream>>>(h, W2bf, b2, x, ab1, h2, stats2);
  finalize_kernel<<<1, 256, 0, stream>>>(stats2, g2, be2, ab2, H_DIM,
                                         1.0f / (float)(T_STEPS * M_ROWS));
  lif2_kernel<<<256, 256, 0, stream>>>(h2, ab2, opre, stats3);
  finalize_kernel<<<1, 256, 0, stream>>>(stats3, g3, be3, ab3, H_DIM,
                                         1.0f / (float)M_ROWS);
  final_kernel<<<(M_ROWS * H_DIM) / 256, 256, 0, stream>>>(opre, ab3, (float*)d_out);
}
